// DJconv_59511066853715
// MI455X (gfx1250) — 8-bit matrix operand order repaired and checked by reading, compiled, not run on hardware
//
#include <hip/hip_runtime.h>

#define KNU 4096
#define KNI 3072

typedef __attribute__((ext_vector_type(8))) int   v8i;
typedef __attribute__((ext_vector_type(8))) float v8f;
typedef __attribute__((ext_vector_type(2))) float v2f;

__global__ __launch_bounds__(256)
void k_binarize(const float* __restrict__ H,
                signed char* __restrict__ H8,
                signed char* __restrict__ Ht8) {
  __shared__ signed char t[32][33];
  int bx = blockIdx.x * 32, by = blockIdx.y * 32;
  int tx = threadIdx.x, ty = threadIdx.y;
#pragma unroll
  for (int r = ty; r < 32; r += 8) {
    int u = by + r, j = bx + tx;
    signed char v = (H[(size_t)u * KNI + j] >= 0.5f) ? 1 : 0;
    H8[(size_t)u * KNI + j] = v;
    t[r][tx] = v;
  }
  __syncthreads();
#pragma unroll
  for (int r = ty; r < 32; r += 8) {
    int j = bx + r, u = by + tx;
    Ht8[(size_t)j * KNU + u] = t[tx][r];
  }
}

__global__ __launch_bounds__(32)
void k_gram_bin(const signed char* __restrict__ X, const signed char* __restrict__ Y,
                signed char* __restrict__ Out, int M, int N, int K) {
  int lane = threadIdx.x;
  int lo = lane & 15, hi = lane >> 4;
  int m0 = blockIdx.y * 32, n0 = blockIdx.x * 32;
  v8i acc[2][2] = {};
  const signed char* ar0 = X + (size_t)(m0 + lo) * K;
  const signed char* ar1 = X + (size_t)(m0 + 16 + lo) * K;
  const signed char* br0 = Y + (size_t)(n0 + lo) * K;
  const signed char* br1 = Y + (size_t)(n0 + 16 + lo) * K;
  for (int k0 = 0; k0 < K; k0 += 64) {
    v8i a[2], b[2];
    const signed char* ap[2] = { ar0 + k0 + hi * 8,  ar1 + k0 + hi * 8 };
    const signed char* bp[2] = { br0 + k0 + hi * 8,  br1 + k0 + hi * 8 };
#pragma unroll
    for (int t = 0; t < 2; ++t) {
#pragma unroll
      for (int q = 0; q < 4; ++q) {
        int2 d = *(const int2*)(ap[t] + q * 16);
        a[t][2 * q] = d.x; a[t][2 * q + 1] = d.y;
      }
#pragma unroll
      for (int q = 0; q < 4; ++q) {
        int2 e = *(const int2*)(bp[t] + q * 16);
        b[t][2 * q] = e.x; b[t][2 * q + 1] = e.y;
      }
    }
#pragma unroll
    for (int mi = 0; mi < 2; ++mi)
#pragma unroll
      for (int ni = 0; ni < 2; ++ni)
        acc[mi][ni] = __builtin_amdgcn_wmma_i32_16x16x64_iu8(
            false, a[mi], false, b[ni], acc[mi][ni], false, false);
  }
#pragma unroll
  for (int mi = 0; mi < 2; ++mi)
#pragma unroll
    for (int ni = 0; ni < 2; ++ni)
#pragma unroll
      for (int v = 0; v < 8; ++v) {
        int m = m0 + mi * 16 + hi * 8 + v;
        int n = n0 + ni * 16 + lo;
        Out[(size_t)m * N + n] = (acc[mi][ni][v] != 0) ? (signed char)1 : (signed char)0;
      }
}

__global__ __launch_bounds__(256)
void k_rowsum_bin(const signed char* __restrict__ A, float* __restrict__ out, int C) {
  __shared__ int red[256];
  int r = blockIdx.x;
  const signed char* row = A + (size_t)r * C;
  int s = 0;
  for (int c = threadIdx.x * 16; c < C; c += 256 * 16) {
    int4 d = *(const int4*)(row + c);
    s += __popc(d.x) + __popc(d.y) + __popc(d.z) + __popc(d.w);
  }
  red[threadIdx.x] = s;
  __syncthreads();
  for (int w = 128; w > 0; w >>= 1) {
    if (threadIdx.x < w) red[threadIdx.x] += red[threadIdx.x + w];
    __syncthreads();
  }
  if (threadIdx.x == 0) out[r] = (float)red[0];
}

__global__ __launch_bounds__(256)
void k_colsum_bin(const signed char* __restrict__ A, float* __restrict__ out, int R, int C) {
  int c = blockIdx.x * 256 + threadIdx.x;
  if (c >= C) return;
  int s = 0;
  for (int r = 0; r < R; ++r) s += A[(size_t)r * C + c];
  out[c] = (float)s;
}

__global__ __launch_bounds__(256)
void k_scales(const float* __restrict__ sRowH, const float* __restrict__ sRowS,
              const float* __restrict__ sColH, const float* __restrict__ sColS,
              float* __restrict__ dvu, float* __restrict__ invRowH, float* __restrict__ invRowS,
              float* __restrict__ dvi, float* __restrict__ invColH, float* __restrict__ invColS) {
  int i = blockIdx.x * 256 + threadIdx.x;
  if (i < KNU) {
    float rv = sRowH[i] + sRowS[i];
    dvu[i] = rv > 0.f ? 1.f / sqrtf(rv) : 0.f;
    invRowH[i] = sRowH[i] > 0.f ? 1.f / sRowH[i] : 0.f;
    invRowS[i] = sRowS[i] > 0.f ? 1.f / sRowS[i] : 0.f;
  }
  if (i < KNI) {
    float cv = sColH[i] + sColS[i];
    dvi[i] = cv > 0.f ? 1.f / sqrtf(cv) : 0.f;
    invColH[i] = sColH[i] > 0.f ? 1.f / sColH[i] : 0.f;
    invColS[i] = sColS[i] > 0.f ? 1.f / sColS[i] : 0.f;
  }
}

__global__ __launch_bounds__(256)
void k_rowscale(const float* __restrict__ X, const float* __restrict__ s,
                float* __restrict__ out, int n) {
  int i = blockIdx.x * 256 + threadIdx.x;
  if (i < n) out[i] = X[i] * s[i >> 6];
}

template <bool TRANS>
__device__ inline void prop_accum(const signed char* __restrict__ Bm, const float* __restrict__ X,
                                  int Klen, int mA, int lo, int hi, int R, v8f acc[4]) {
  for (int k0 = 0; k0 < Klen; k0 += 4) {
    int ka = k0 + hi * 2;
    float a0, a1;
    if (TRANS) {
      a0 = (float)Bm[(size_t)ka * R + mA];
      a1 = (float)Bm[(size_t)(ka + 1) * R + mA];
    } else {
      a0 = (float)Bm[(size_t)mA * Klen + ka];
      a1 = (float)Bm[(size_t)mA * Klen + ka + 1];
    }
    v2f a; a[0] = a0; a[1] = a1;
#pragma unroll
    for (int ni = 0; ni < 4; ++ni) {
      int n = ni * 16 + lo;
      v2f b;
      b[0] = X[(size_t)ka * 64 + n];
      b[1] = X[(size_t)(ka + 1) * 64 + n];
      acc[ni] = __builtin_amdgcn_wmma_f32_16x16x4_f32(
          false, a, false, b, (short)0, acc[ni], false, false);
    }
  }
}

template <bool TRANS>
__global__ __launch_bounds__(32)
void k_prop_f32(const signed char* __restrict__ B1, const float* __restrict__ X1, int K1,
                const signed char* __restrict__ B2, const float* __restrict__ X2, int K2,
                const float* __restrict__ oscale, const float* __restrict__ addin,
                float* __restrict__ out, int R) {
  int lane = threadIdx.x;
  int lo = lane & 15, hi = lane >> 4;
  int m0 = blockIdx.x * 16;
  int mA = m0 + lo;
  v8f acc[4] = {};
  prop_accum<TRANS>(B1, X1, K1, mA, lo, hi, R, acc);
  if (B2) prop_accum<TRANS>(B2, X2, K2, mA, lo, hi, R, acc);
#pragma unroll
  for (int ni = 0; ni < 4; ++ni)
#pragma unroll
    for (int v = 0; v < 8; ++v) {
      int m = m0 + hi * 8 + v;
      int n = ni * 16 + lo;
      float r = acc[ni][v];
      if (oscale) r *= oscale[m];
      if (addin) r += addin[(size_t)m * 64 + n];
      out[(size_t)m * 64 + n] = r;
    }
}

__global__ __launch_bounds__(256)
void k_linear(const float* __restrict__ Y, const float* __restrict__ W,
              const float* __restrict__ bias, float* __restrict__ out, int R) {
  int idx = blockIdx.x * 256 + threadIdx.x;
  if (idx >= R * 64) return;
  int r = idx >> 6, d = idx & 63;
  float s = bias[d];
#pragma unroll 8
  for (int k = 0; k < 64; ++k) s = fmaf(Y[(size_t)r * 64 + k], W[(size_t)k * 64 + d], s);
  out[idx] = s;
}

extern "C" void kernel_launch(void* const* d_in, const int* in_sizes, int n_in,
                              void* d_out, int out_size, void* d_ws, size_t ws_size,
                              hipStream_t stream) {
  const float* H    = (const float*)d_in[0];
  const float* U    = (const float*)d_in[1];
  const float* I    = (const float*)d_in[2];
  const float* W    = (const float*)d_in[3];
  const float* bias = (const float*)d_in[4];
  float* Uout = (float*)d_out;
  float* Iout = Uout + (size_t)KNU * 64;

  size_t off = 0;
  auto wsalloc = [&](size_t bytes) -> char* {
    char* p = (char*)d_ws + off;
    off += (bytes + 255) & ~(size_t)255;
    return p;
  };
  signed char* H8   = (signed char*)wsalloc((size_t)KNU * KNI);
  signed char* Ht8  = (signed char*)wsalloc((size_t)KNI * KNU);
  signed char* Gb   = (signed char*)wsalloc((size_t)KNI * KNI);
  signed char* S    = (signed char*)wsalloc((size_t)KNU * KNI);
  float* sRowH   = (float*)wsalloc(KNU * 4);
  float* sColH   = (float*)wsalloc(KNI * 4);
  float* sRowS   = (float*)wsalloc(KNU * 4);
  float* sColS   = (float*)wsalloc(KNI * 4);
  float* dvu     = (float*)wsalloc(KNU * 4);
  float* invRowH = (float*)wsalloc(KNU * 4);
  float* invRowS = (float*)wsalloc(KNU * 4);
  float* dvi     = (float*)wsalloc(KNI * 4);
  float* invColH = (float*)wsalloc(KNI * 4);
  float* invColS = (float*)wsalloc(KNI * 4);
  float* Xsu  = (float*)wsalloc((size_t)KNU * 64 * 4);
  float* Xsi  = (float*)wsalloc((size_t)KNI * 64 * 4);
  float* Q1   = (float*)wsalloc((size_t)KNI * 64 * 4);
  float* Q2   = (float*)wsalloc((size_t)KNI * 64 * 4);
  float* Qi1  = (float*)wsalloc((size_t)KNU * 64 * 4);
  float* Qi2  = (float*)wsalloc((size_t)KNU * 64 * 4);
  float* Yu   = (float*)wsalloc((size_t)KNU * 64 * 4);
  float* Yi   = (float*)wsalloc((size_t)KNI * 64 * 4);
  (void)ws_size; (void)in_sizes; (void)n_in; (void)out_size;

  k_binarize<<<dim3(KNI / 32, KNU / 32), dim3(32, 8), 0, stream>>>(H, H8, Ht8);
  k_gram_bin<<<dim3(KNI / 32, KNI / 32), 32, 0, stream>>>(Ht8, Ht8, Gb, KNI, KNI, KNU);
  k_gram_bin<<<dim3(KNI / 32, KNU / 32), 32, 0, stream>>>(H8, Gb, S, KNU, KNI, KNI);

  k_rowsum_bin<<<KNU, 256, 0, stream>>>(H8, sRowH, KNI);
  k_rowsum_bin<<<KNI, 256, 0, stream>>>(Ht8, sColH, KNU);
  k_rowsum_bin<<<KNU, 256, 0, stream>>>(S, sRowS, KNI);
  k_colsum_bin<<<KNI / 256, 256, 0, stream>>>(S, sColS, KNU, KNI);
  k_scales<<<(KNU + 255) / 256, 256, 0, stream>>>(sRowH, sRowS, sColH, sColS,
                                                  dvu, invRowH, invRowS,
                                                  dvi, invColH, invColS);

  k_rowscale<<<(KNU * 64) / 256, 256, 0, stream>>>(U, dvu, Xsu, KNU * 64);
  k_prop_f32<true><<<KNI / 16, 32, 0, stream>>>(H8, Xsu, KNU, nullptr, nullptr, 0,
                                                invColH, nullptr, Q1, KNI);
  k_prop_f32<true><<<KNI / 16, 32, 0, stream>>>(S, Xsu, KNU, nullptr, nullptr, 0,
                                                invColS, nullptr, Q2, KNI);
  k_prop_f32<false><<<KNU / 16, 32, 0, stream>>>(H8, Q1, KNI, S, Q2, KNI,
                                                 dvu, U, Yu, KNU);

  k_rowscale<<<(KNI * 64) / 256, 256, 0, stream>>>(I, dvi, Xsi, KNI * 64);
  k_prop_f32<false><<<KNU / 16, 32, 0, stream>>>(H8, Xsi, KNI, nullptr, nullptr, 0,
                                                 invRowH, nullptr, Qi1, KNU);
  k_prop_f32<false><<<KNU / 16, 32, 0, stream>>>(S, Xsi, KNI, nullptr, nullptr, 0,
                                                 invRowS, nullptr, Qi2, KNU);
  k_prop_f32<true><<<KNI / 16, 32, 0, stream>>>(H8, Qi1, KNU, S, Qi2, KNU,
                                                dvi, I, Yi, KNI);

  k_linear<<<(KNU * 64) / 256, 256, 0, stream>>>(Yu, W, bias, Uout, KNU);
  k_linear<<<(KNI * 64) / 256, 256, 0, stream>>>(Yi, W, bias, Iout, KNI);
}
